// ChunkedTokenMemoryPartialLM_34368328303147
// MI455X (gfx1250) — matrix operand order repaired and checked by reading, not compiled, not run on hardware
//
#include <hip/hip_runtime.h>
#include <hip/hip_bf16.h>
#include <math.h>

// Problem dims
#define S_LEN 2048
#define VOCAB 50257
#define EDIM 512
#define HDIM 768
#define MDIM 256
#define CSIZE 64
#define UDIM 2048
#define NCHUNK 32
#define H3 2304
#define E4 2048

typedef __attribute__((ext_vector_type(16))) __bf16 v16bf;
typedef __attribute__((ext_vector_type(8)))  float  v8f;
typedef __attribute__((ext_vector_type(4)))  unsigned int uint4v;
typedef __attribute__((ext_vector_type(4)))  unsigned int u32x4;
typedef __attribute__((ext_vector_type(8)))  int i32x8;
typedef __attribute__((ext_vector_type(4)))  int i32x4;

union V16U { uint4v u4[2]; v16bf b; };

__device__ __forceinline__ int imin(int a, int b) { return a < b ? a : b; }

// ---------------- TDM: 2D bf16 tile Global -> LDS ----------------
// D# per CDNA5 ISA ch.8: group0 = {count, lds_addr, global_addr, type=2},
// group1 = {data_size=2B, tensor_dim0/1 (relative to tile start, clamps edges
// via OOB-returns-zero), tile_dim0/1, tensor_dim0_stride}.
__device__ __forceinline__ void tdm_load_tile_2d(
    const void* gtile, unsigned lds_byte_off,
    int rows_remaining, int cols_remaining,
    int row_stride_elems, int tile_rows, int tile_cols) {
  unsigned long long ga = (unsigned long long)(uintptr_t)gtile;
  u32x4 g0;
  g0[0] = 1u;                                                   // count=1 (valid), user mode
  g0[1] = lds_byte_off;                                         // lds_addr [63:32]
  g0[2] = (unsigned)(ga & 0xFFFFFFFFu);                         // global_addr[31:0]
  g0[3] = (unsigned)((ga >> 32) & 0x01FFFFFFu) | (2u << 30);    // global_addr[56:32] | type=2
  i32x8 g1;
  g1[0] = (int)(1u << 16);                                      // data_size=1 -> 2 bytes
  g1[1] = (int)(((unsigned)cols_remaining & 0xFFFFu) << 16);    // tensor_dim0[15:0]
  g1[2] = (int)(((unsigned)cols_remaining >> 16) |
                (((unsigned)rows_remaining & 0xFFFFu) << 16));  // dim0[31:16] | dim1[15:0]
  g1[3] = (int)(((unsigned)rows_remaining >> 16) |
                ((unsigned)tile_cols << 16));                   // dim1[31:16] | tile_dim0
  g1[4] = (int)((unsigned)tile_rows & 0xFFFFu);                 // tile_dim1 (tile_dim2=0)
  g1[5] = row_stride_elems;                                     // tensor_dim0_stride[31:0]
  g1[6] = 0;                                                    // stride hi / dim1_stride lo
  g1[7] = 0;                                                    // dim1_stride hi
  i32x4 z4 = {0, 0, 0, 0};
#if defined(__clang_major__) && (__clang_major__ >= 23)
  i32x8 z8 = {0, 0, 0, 0, 0, 0, 0, 0};
  __builtin_amdgcn_tensor_load_to_lds(g0, g1, z4, z4, z8, 0);
#else
  __builtin_amdgcn_tensor_load_to_lds(g0, g1, z4, z4, 0);
#endif
}

// ---------------- WMMA fragment loads from LDS ----------------
// A tile [128 rows(M)][32 cols(K)] bf16 row-major.
// ISA 16-bit A 16x32: lane L<16 -> M=L, K={0..7,16..23}; L>=16 -> M=L-16, K={8..15,24..31}.
__device__ __forceinline__ v16bf load_a_frag(const __bf16* As, int mrow, int lane) {
  int m  = mrow + (lane & 15);
  int c0 = (lane >> 4) * 8;
  const __bf16* p = As + m * 32 + c0;
  V16U r;
  r.u4[0] = *(const uint4v*)(p);
  r.u4[1] = *(const uint4v*)(p + 16);
  return r.b;
}
// B tile [128 rows(N)][32 cols(K)] bf16: lane holds N=lane&15, K={0..7,16..23} (L<16) or {8..15,24..31} (L>=16).
__device__ __forceinline__ v16bf load_b_frag(const __bf16* Bs, int nrow, int lane) {
  int n  = nrow + (lane & 15);
  int kb = (lane >> 4) * 8;
  const __bf16* p = Bs + n * 32 + kb;
  V16U r;
  r.u4[0] = *(const uint4v*)(p);
  r.u4[1] = *(const uint4v*)(p + 16);
  return r.b;
}

// ---------------- Generic NT GEMM: C[M,N] = A[M,K] @ B[N,K]^T + bias ----------------
// Double-buffered LDS. A tile staged by TDM (TENSORcnt), B tile by
// GLOBAL_LOAD_ASYNC_TO_LDS_B128 (ASYNCcnt); copy of tile k+1 overlaps WMMA on tile k.
// ACT: 0 = none, 1 = relu(x)^2.  OUTBF: 0 = fp32 out, 1 = bf16 out.
template <int ACT, int OUTBF>
__global__ __launch_bounds__(256) void gemm_bf16_kernel(
    const __bf16* __restrict__ A, const __bf16* __restrict__ B,
    const float* __restrict__ bias, void* __restrict__ Cout,
    int Md, int Nd, int Kd) {
  __shared__ __bf16 As[2][128 * 32];
  __shared__ __bf16 Bs[2][128 * 32];
  const int tid    = threadIdx.x;
  const int lane   = tid & 31;
  const int wave   = tid >> 5;
  const int wm     = wave >> 2;  // 0..1  (64-row slab)
  const int wn     = wave & 3;   // 0..3  (32-col slab)
  const int blockN = blockIdx.x * 128;
  const int blockM = blockIdx.y * 128;

  v8f acc[4][2] = {};

  // B staging: each thread copies 16B (8 bf16); 4 threads per 32-elem row.
  const int srow = tid >> 2;        // 0..63
  const int scol = (tid & 3) * 8;   // 0,8,16,24
  const unsigned asBase0 = (unsigned)(uintptr_t)&As[0][0];
  const unsigned asBase1 = (unsigned)(uintptr_t)&As[1][0];
  const unsigned bsBase0 = (unsigned)(uintptr_t)&Bs[0][0];
  const unsigned bsBase1 = (unsigned)(uintptr_t)&Bs[1][0];

  auto stage_b = [&](int buf, int k0) {
    unsigned base = buf ? bsBase1 : bsBase0;
#pragma unroll
    for (int pass = 0; pass < 2; ++pass) {
      int r = pass * 64 + srow;
      int gb = imin(blockN + r, Nd - 1);  // clamp; stores guarded below
      const __bf16* gpb = B + (size_t)gb * Kd + k0 + scol;
      unsigned lb = base + (unsigned)(r * 32 + scol) * 2u;
      asm volatile("global_load_async_to_lds_b128 %0, %1, off"
                   :: "v"(lb), "v"(gpb) : "memory");
    }
  };
  auto stage_a = [&](int buf, int k0) {
    tdm_load_tile_2d(A + (size_t)blockM * Kd + k0, buf ? asBase1 : asBase0,
                     Md - blockM, Kd - k0, Kd, 128, 32);
  };

  // Prologue: stage tile 0 into buffer 0
  stage_b(0, 0);
  if (wave == 0) stage_a(0, 0);
  asm volatile("s_wait_asynccnt 0x0" ::: "memory");
  if (wave == 0) __builtin_amdgcn_s_wait_tensorcnt(0);
  __syncthreads();

  const int nk = Kd / 32;
  for (int ik = 0; ik < nk; ++ik) {
    const int cur = ik & 1;
    const int k0n = (ik + 1) * 32;
    if (k0n < Kd) {  // kick off next tile's copies (overlap with WMMA below)
      stage_b(cur ^ 1, k0n);
      if (wave == 0) stage_a(cur ^ 1, k0n);
      if (k0n + 32 < Kd) {
        __builtin_prefetch(A + (size_t)imin(blockM + srow, Md - 1) * Kd + k0n + 32 + scol, 0, 3);
        __builtin_prefetch(B + (size_t)imin(blockN + srow, Nd - 1) * Kd + k0n + 32 + scol, 0, 3);
      }
    }
    const __bf16* Asc = As[cur];
    const __bf16* Bsc = Bs[cur];
    v16bf b0 = load_b_frag(Bsc, wn * 32,      lane);
    v16bf b1 = load_b_frag(Bsc, wn * 32 + 16, lane);
#pragma unroll
    for (int mi = 0; mi < 4; ++mi) {
      v16bf a = load_a_frag(Asc, wm * 64 + mi * 16, lane);
      acc[mi][0] = __builtin_amdgcn_wmma_f32_16x16x32_bf16(
          false, a, false, b0, (short)0, acc[mi][0], false, false);
      acc[mi][1] = __builtin_amdgcn_wmma_f32_16x16x32_bf16(
          false, a, false, b1, (short)0, acc[mi][1], false, false);
    }
    __syncthreads();  // all waves done reading buf[cur]
    if (k0n < Kd) {   // publish buf[cur^1] before next iteration reads it
      asm volatile("s_wait_asynccnt 0x0" ::: "memory");
      if (wave == 0) __builtin_amdgcn_s_wait_tensorcnt(0);
      __syncthreads();
    }
  }

  // C/D layout: VGPR r -> M = r + 8*(lane>=16), N = lane&15
  const int rsel = (lane >> 4) * 8;
  const int csel = lane & 15;
#pragma unroll
  for (int mi = 0; mi < 4; ++mi) {
#pragma unroll
    for (int ni = 0; ni < 2; ++ni) {
      int col = blockN + wn * 32 + ni * 16 + csel;
      if (col < Nd) {
        float bv = bias[col];
#pragma unroll
        for (int r = 0; r < 8; ++r) {
          int row = blockM + wm * 64 + mi * 16 + rsel + r;
          if (row < Md) {
            float v = acc[mi][ni][r] + bv;
            if (ACT == 1) { v = fmaxf(v, 0.f); v = v * v; }
            if (OUTBF) ((__bf16*)Cout)[(size_t)row * Nd + col] = (__bf16)v;
            else       ((float*)Cout)[(size_t)row * Nd + col] = v;
          }
        }
      }
    }
  }
}

// ---------------- fp32 -> bf16 cast ----------------
__global__ void cast_f32_bf16_kernel(const float* __restrict__ in,
                                     __bf16* __restrict__ out, long n) {
  long i = ((long)blockIdx.x * blockDim.x + threadIdx.x) * 8;
  if (i + 8 <= n) {
    float4 a = *(const float4*)(in + i);
    float4 b = *(const float4*)(in + i + 4);
    union { __bf16 h[8]; uint4v u; } t;
    t.h[0] = (__bf16)a.x; t.h[1] = (__bf16)a.y; t.h[2] = (__bf16)a.z; t.h[3] = (__bf16)a.w;
    t.h[4] = (__bf16)b.x; t.h[5] = (__bf16)b.y; t.h[6] = (__bf16)b.z; t.h[7] = (__bf16)b.w;
    *(uint4v*)(out + i) = t.u;
  } else {
    for (; i < n; ++i) out[i] = (__bf16)in[i];
  }
}

// ---------------- Embedding gather (bf16 out) ----------------
__global__ void gather_kernel(const int* __restrict__ ids,
                              const float* __restrict__ emb,
                              __bf16* __restrict__ x) {
  int s  = blockIdx.x;
  int id = ids[s];
  for (int e = threadIdx.x; e < EDIM; e += blockDim.x)
    x[(size_t)s * EDIM + e] = (__bf16)emb[(size_t)id * EDIM + e];
}

// ---------------- GRU recurrence (gx precomputed) ----------------
__global__ __launch_bounds__(1024) void gru_kernel(
    const float* __restrict__ gx, const float* __restrict__ w_hh,
    const float* __restrict__ b_hh, float* __restrict__ states,
    __bf16* __restrict__ states_bf) {
  __shared__ float h_s[HDIM];
  __shared__ float gh_s[H3];
  const int tid = threadIdx.x;
  for (int i = tid; i < HDIM; i += 1024) h_s[i] = 0.f;
  __syncthreads();
  for (int t = 0; t < S_LEN; ++t) {
    for (int o = tid; o < H3; o += 1024) {
      const float* wr = w_hh + (size_t)o * HDIM;
      float acc = b_hh[o];
#pragma unroll 4
      for (int k = 0; k < HDIM; ++k) acc += wr[k] * h_s[k];
      gh_s[o] = acc;
    }
    __syncthreads();
    const float* gxt = gx + (size_t)t * H3;
    for (int i = tid; i < HDIM; i += 1024) {
      float r  = 1.f / (1.f + __expf(-(gxt[i]            + gh_s[i])));
      float z  = 1.f / (1.f + __expf(-(gxt[HDIM + i]     + gh_s[HDIM + i])));
      float n  = tanhf(gxt[2 * HDIM + i] + r * gh_s[2 * HDIM + i]);
      float hn = (1.f - z) * n + z * h_s[i];
      h_s[i] = hn;
      states[(size_t)t * HDIM + i] = hn;
      states_bf[(size_t)t * HDIM + i] = (__bf16)hn;
    }
    __syncthreads();
  }
}

// ---------------- Chunk means (bf16 out) ----------------
__global__ void chunk_mean_kernel(const float* __restrict__ states,
                                  __bf16* __restrict__ chunked) {
  int c = blockIdx.x;
  for (int i = threadIdx.x; i < HDIM; i += blockDim.x) {
    float s = 0.f;
    for (int j = 0; j < CSIZE; ++j)
      s += states[((size_t)(c * CSIZE + j)) * HDIM + i];
    chunked[c * HDIM + i] = (__bf16)(s * (1.0f / (float)CSIZE));
  }
}

// ---------------- Attention over chunks + scatter-add into logits ----------------
__global__ __launch_bounds__(256) void attn_scatter_kernel(
    const float* __restrict__ qk, const float* __restrict__ mk,
    const float* __restrict__ base_partial, const float* __restrict__ chunk_partial,
    const int* __restrict__ untied, float* __restrict__ out) {
  __shared__ float q_s[MDIM];
  __shared__ float attn_s[NCHUNK];
  const int s   = blockIdx.x;
  const int tid = threadIdx.x;
  q_s[tid] = qk[(size_t)s * MDIM + tid];
  __syncthreads();
  if (tid < NCHUNK) {
    int c = tid;
    float v = -3.0e38f;
    if (c * CSIZE + (CSIZE - 1) < s) {
      float d = 0.f;
      const float* mkr = mk + c * MDIM;
      for (int m = 0; m < MDIM; ++m) d += q_s[m] * mkr[m];
      v = d * 0.0625f;  // 1/sqrt(256)
    }
    attn_s[c] = v;
  }
  __syncthreads();
  if (tid == 0) {
    float mx = -3.0e38f;
    for (int c = 0; c < NCHUNK; ++c) mx = fmaxf(mx, attn_s[c]);
    if (mx > -1.0e38f) {
      float sum = 0.f;
      for (int c = 0; c < NCHUNK; ++c) {
        float e = (attn_s[c] > -1.0e38f) ? __expf(attn_s[c] - mx) : 0.f;
        attn_s[c] = e; sum += e;
      }
      float inv = 1.f / fmaxf(sum, 1e-6f);
      for (int c = 0; c < NCHUNK; ++c) attn_s[c] *= inv;
    } else {
      for (int c = 0; c < NCHUNK; ++c) attn_s[c] = 0.f;
    }
  }
  __syncthreads();
  for (int u = tid; u < UDIM; u += 256) {
    float acc = base_partial[(size_t)s * UDIM + u];
    for (int c = 0; c < NCHUNK; ++c) acc += attn_s[c] * chunk_partial[c * UDIM + u];
    atomicAdd(out + (size_t)s * VOCAB + untied[u], acc);
  }
}

// ---------------- Host-side orchestration ----------------
static void launch_cast(const float* in, __bf16* out, long n, hipStream_t s) {
  long blocks = (n / 8 + 255) / 256 + 1;
  cast_f32_bf16_kernel<<<dim3((unsigned)blocks), 256, 0, s>>>(in, out, n);
}

extern "C" void kernel_launch(void* const* d_in, const int* in_sizes, int n_in,
                              void* d_out, int out_size, void* d_ws, size_t ws_size,
                              hipStream_t stream) {
  const int*   input_ids = (const int*)d_in[0];
  const int*   untied    = (const int*)d_in[1];
  const float* emb   = (const float*)d_in[2];
  const float* w_ih  = (const float*)d_in[3];
  const float* w_hh  = (const float*)d_in[4];
  const float* b_ih  = (const float*)d_in[5];
  const float* b_hh  = (const float*)d_in[6];
  const float* Wq    = (const float*)d_in[7];
  const float* bq    = (const float*)d_in[8];
  const float* Wk    = (const float*)d_in[9];
  const float* bk    = (const float*)d_in[10];
  const float* Whf   = (const float*)d_in[11];
  const float* bhf   = (const float*)d_in[12];
  const float* Whp   = (const float*)d_in[13];
  const float* bhp   = (const float*)d_in[14];
  const float* Wcv   = (const float*)d_in[15];
  const float* bcv   = (const float*)d_in[16];
  const float* Wph   = (const float*)d_in[17];
  const float* bph   = (const float*)d_in[18];
  const float* out_b = (const float*)d_in[19];
  float* out = (float*)d_out;

  char* w = (char*)d_ws;
  auto alloc = [&](size_t bytes) -> char* {
    char* r = w; w += (bytes + 255) & ~(size_t)255; return r;
  };
  __bf16* emb_bf   = (__bf16*)alloc((size_t)VOCAB * EDIM * 2);
  __bf16* wih_bf   = (__bf16*)alloc((size_t)H3 * EDIM * 2);
  __bf16* whf_bf   = (__bf16*)alloc((size_t)E4 * HDIM * 2);
  __bf16* whp_bf   = (__bf16*)alloc((size_t)EDIM * E4 * 2);
  __bf16* wq_bf    = (__bf16*)alloc((size_t)MDIM * HDIM * 2);
  __bf16* wk_bf    = (__bf16*)alloc((size_t)MDIM * HDIM * 2);
  __bf16* wcv_bf   = (__bf16*)alloc((size_t)EDIM * HDIM * 2);
  __bf16* wph_bf   = (__bf16*)alloc((size_t)UDIM * EDIM * 2);
  __bf16* x_bf     = (__bf16*)alloc((size_t)S_LEN * EDIM * 2);
  float*  gx       = (float*) alloc((size_t)S_LEN * H3 * 4);
  float*  states   = (float*) alloc((size_t)S_LEN * HDIM * 4);
  __bf16* states_bf= (__bf16*)alloc((size_t)S_LEN * HDIM * 2);
  __bf16* head_bf  = (__bf16*)alloc((size_t)S_LEN * E4 * 2);
  __bf16* bfeat_bf = (__bf16*)alloc((size_t)S_LEN * EDIM * 2);
  float*  qk       = (float*) alloc((size_t)S_LEN * MDIM * 4);
  __bf16* chunk_bf = (__bf16*)alloc((size_t)NCHUNK * HDIM * 2);
  float*  mk       = (float*) alloc((size_t)NCHUNK * MDIM * 4);
  __bf16* cv_bf    = (__bf16*)alloc((size_t)NCHUNK * EDIM * 2);
  float*  chunk_pl = (float*) alloc((size_t)NCHUNK * UDIM * 4);
  float*  base_pl  = (float*) alloc((size_t)S_LEN * UDIM * 4);

  // 0) weight precasts to bf16 (L2-friendly; halves global traffic for GEMMs)
  launch_cast(emb,  emb_bf, (long)VOCAB * EDIM, stream);
  launch_cast(w_ih, wih_bf, (long)H3 * EDIM, stream);
  launch_cast(Whf,  whf_bf, (long)E4 * HDIM, stream);
  launch_cast(Whp,  whp_bf, (long)EDIM * E4, stream);
  launch_cast(Wq,   wq_bf,  (long)MDIM * HDIM, stream);
  launch_cast(Wk,   wk_bf,  (long)MDIM * HDIM, stream);
  launch_cast(Wcv,  wcv_bf, (long)EDIM * HDIM, stream);
  launch_cast(Wph,  wph_bf, (long)UDIM * EDIM, stream);

  // 1) x = emb[input_ids]  (bf16)
  gather_kernel<<<S_LEN, 256, 0, stream>>>(input_ids, emb, x_bf);
  // 2) gx = x @ w_ih^T + b_ih  (fp32 out, feeds GRU)
  { dim3 g((H3 + 127) / 128, (S_LEN + 127) / 128);
    gemm_bf16_kernel<0,0><<<g, 256, 0, stream>>>(x_bf, wih_bf, b_ih, gx, S_LEN, H3, EDIM); }
  // 3) GRU recurrence -> states fp32 + bf16
  gru_kernel<<<1, 1024, 0, stream>>>(gx, w_hh, b_hh, states, states_bf);
  // 4) head = relu(states @ Whf^T + bhf)^2  (bf16 out)
  { dim3 g((E4 + 127) / 128, (S_LEN + 127) / 128);
    gemm_bf16_kernel<1,1><<<g, 256, 0, stream>>>(states_bf, whf_bf, bhf, head_bf, S_LEN, E4, HDIM); }
  // 5) base_feat = head @ Whp^T + bhp  (bf16 out)
  { dim3 g((EDIM + 127) / 128, (S_LEN + 127) / 128);
    gemm_bf16_kernel<0,1><<<g, 256, 0, stream>>>(head_bf, whp_bf, bhp, bfeat_bf, S_LEN, EDIM, E4); }
  // 6) qk = states @ Wq^T + bq  (fp32 out)
  { dim3 g((MDIM + 127) / 128, (S_LEN + 127) / 128);
    gemm_bf16_kernel<0,0><<<g, 256, 0, stream>>>(states_bf, wq_bf, bq, qk, S_LEN, MDIM, HDIM); }
  // 7) chunk means (bf16 out)
  chunk_mean_kernel<<<NCHUNK, 256, 0, stream>>>(states, chunk_bf);
  // 8) mk = chunked @ Wk^T + bk  (fp32 out)
  { dim3 g((MDIM + 127) / 128, 1);
    gemm_bf16_kernel<0,0><<<g, 256, 0, stream>>>(chunk_bf, wk_bf, bk, mk, NCHUNK, MDIM, HDIM); }
  // 9) cv = chunked @ Wcv^T + bcv  (bf16 out)
  { dim3 g((EDIM + 127) / 128, 1);
    gemm_bf16_kernel<0,1><<<g, 256, 0, stream>>>(chunk_bf, wcv_bf, bcv, cv_bf, NCHUNK, EDIM, HDIM); }
  // 10) chunk_partial = cv @ Wph^T + bph  (fp32 out)
  { dim3 g((UDIM + 127) / 128, 1);
    gemm_bf16_kernel<0,0><<<g, 256, 0, stream>>>(cv_bf, wph_bf, bph, chunk_pl, NCHUNK, UDIM, EDIM); }
  // 11) base_partial = base_feat @ Wph^T + bph  (fp32 out)
  { dim3 g((UDIM + 127) / 128, (S_LEN + 127) / 128);
    gemm_bf16_kernel<0,0><<<g, 256, 0, stream>>>(bfeat_bf, wph_bf, bph, base_pl, S_LEN, UDIM, EDIM); }
  // 12) base_logits = base_feat @ emb^T + out_bias  (direct to d_out)
  { dim3 g((VOCAB + 127) / 128, (S_LEN + 127) / 128);
    gemm_bf16_kernel<0,0><<<g, 256, 0, stream>>>(bfeat_bf, emb_bf, out_b, out, S_LEN, VOCAB, EDIM); }
  // 13) attention over chunks + scatter-add partial logits
  attn_scatter_kernel<<<S_LEN, 256, 0, stream>>>(qk, mk, base_pl, chunk_pl, untied, out);
}
